// GATConv_15247133900894
// MI455X (gfx1250) — matrix operand order repaired and checked by reading, not compiled, not run on hardware
//
#include <hip/hip_runtime.h>
#include <hip/hip_bf16.h>

// ---- sizes (fixed by the reference) ----
#define NN    8192
#define FIN   256
#define FOUT  128
#define NEG   0.2f

typedef __attribute__((ext_vector_type(2)))  float        v2f;
typedef __attribute__((ext_vector_type(8)))  float        v8f;
typedef __attribute__((ext_vector_type(16))) __bf16       v16bf;
typedef __attribute__((ext_vector_type(4)))  unsigned int v4u;
typedef int v4i_g __attribute__((vector_size(16)));   // matches builtin param

#define AS1 __attribute__((address_space(1)))
#define AS3 __attribute__((address_space(3)))

// ============================================================================
// Kernel A: h = x @ W  (fp32 WMMA 16x16x4), also emit transposed bf16 copy
// ============================================================================
__global__ void __launch_bounds__(256) gemm1_xw(const float* __restrict__ x,
                                                const float* __restrict__ W,
                                                float* __restrict__ h,
                                                __bf16* __restrict__ ht) {
  const int mt  = blockIdx.x;          // 0..511 : 16-row tile
  const int nt  = threadIdx.x >> 5;    // 0..7   : 16-col tile
  const int L   = threadIdx.x & 31;
  const int lh  = L >> 4;              // lane half
  const int l15 = L & 15;

  v8f acc = {0.f,0.f,0.f,0.f,0.f,0.f,0.f,0.f};
  const int rowA = mt * 16 + l15;
  const int colB = nt * 16 + l15;

  for (int k0 = 0; k0 < FIN; k0 += 4) {
    v2f a = *(const v2f*)(x + (size_t)rowA * FIN + (k0 + lh * 2));
    v2f b;
    b.x = W[(size_t)(k0 + lh * 2 + 0) * FOUT + colB];
    b.y = W[(size_t)(k0 + lh * 2 + 1) * FOUT + colB];
    acc = __builtin_amdgcn_wmma_f32_16x16x4_f32(false, a, false, b,
                                                (short)0, acc, false, false);
  }
#pragma unroll
  for (int v = 0; v < 8; ++v) {
    int row = mt * 16 + v + lh * 8;
    float val = acc[v];
    h[(size_t)row * FOUT + colB] = val;
    ht[(size_t)colB * NN + row]  = (__bf16)val;   // transposed bf16 copy
  }
}

// ============================================================================
// Kernel B: s_src = h@a1, s_dst = h@a2   (wave per row, shuffle reduce)
// ============================================================================
__global__ void __launch_bounds__(256) scores(const float* __restrict__ h,
                                              const float* __restrict__ a1,
                                              const float* __restrict__ a2,
                                              float* __restrict__ ssrc,
                                              float* __restrict__ sdst) {
  int row = (int)((blockIdx.x * blockDim.x + threadIdx.x) >> 5);
  int L = threadIdx.x & 31;
  float4 hv = ((const float4*)(h + (size_t)row * FOUT))[L];
  float4 av = ((const float4*)a1)[L];
  float4 bv = ((const float4*)a2)[L];
  float s1 = hv.x*av.x + hv.y*av.y + hv.z*av.z + hv.w*av.w;
  float s2 = hv.x*bv.x + hv.y*bv.y + hv.z*bv.z + hv.w*bv.w;
#pragma unroll
  for (int off = 16; off > 0; off >>= 1) {
    s1 += __shfl_xor(s1, off, 32);
    s2 += __shfl_xor(s2, off, 32);
  }
  if (L == 0) { ssrc[row] = s1; sdst[row] = s2; }
}

// ============================================================================
// Kernel C: pack adjacency to bitmask (256 MB read once -> 8 MB, L2-resident)
// ============================================================================
__global__ void __launch_bounds__(256) packmask(const int* __restrict__ adj,
                                                unsigned int* __restrict__ mb) {
  size_t gid = (size_t)blockIdx.x * 256 + threadIdx.x;
  unsigned int m = __builtin_amdgcn_ballot_w32(adj[gid] > 0);
  if ((threadIdx.x & 31) == 0) mb[gid >> 5] = m;
}

// ============================================================================
// Kernel D: fused masked softmax + (attention @ h) with bf16 WMMA.
//   block = 256 threads (8 waves) handles 32 output rows (2 M-tiles); wave w
//   owns cols [16w,16w+16) and carries TWO accumulators so every B-fragment
//   load feeds two WMMAs (halves L2 traffic on ht).
//   Deferred-normalizer softmax: O_unnorm and Z accumulated, out = O/Z + bias.
//   grid = 256 blocks.
// ============================================================================
__global__ void __launch_bounds__(256) gat_softmax_agg(
    const __bf16* __restrict__ ht,          // [FOUT][NN] transposed bf16 h
    const unsigned int* __restrict__ mb,    // [NN][NN/32] adjacency bits
    const float* __restrict__ ssrc,
    const float* __restrict__ sdst,
    const float* __restrict__ bias,
    float* __restrict__ out) {
  __shared__ float sdl[NN];                               // 32 KB: s_dst copy
  __shared__ float red[512];
  __shared__ float ssl[32];
  __shared__ float mrow[32];
  __shared__ float Zl[32];
  __shared__ alignas(32) unsigned short ptile[32 * 256];  // 16 KB bf16 P tile

  const int t    = threadIdx.x;
  const int row0 = blockIdx.x * 32;

  // ---- stage s_dst (32 KB) into LDS: async-to-LDS path when available ----
#if __has_builtin(__builtin_amdgcn_global_load_async_to_lds_b128) && \
    __has_builtin(__builtin_amdgcn_s_wait_asynccnt)
  {
#pragma unroll
    for (int i = 0; i < 8; ++i) {
      int e = (t + i * 256) * 4;                          // float index, 16B units
      __builtin_amdgcn_global_load_async_to_lds_b128(
          (AS1 v4i_g*)(sdst + e), (AS3 v4i_g*)(sdl + e), 0, 0);
    }
    __builtin_amdgcn_s_wait_asynccnt(0);
  }
#else
  for (int i = t; i < NN / 4; i += 256)
    ((float4*)sdl)[i] = ((const float4*)sdst)[i];
#endif
  __syncthreads();

  // ---- phase 1: per-row max of s_dst over neighbors (leakyrelu monotone) ----
  {
    int r = t & 31, c = t >> 5;                 // 8 chunks of 1024 cols per row
    const unsigned int* mw = mb + (size_t)(row0 + r) * (NN / 32) + c * 32;
    float md = -3.0e38f;
    for (int i = 0; i < 32; ++i) {
      unsigned int word = mw[i];
      int jb = c * 1024 + i * 32;
#pragma unroll
      for (int b = 0; b < 32; ++b) {
        float sv = sdl[jb + b];
        if (word & (1u << b)) md = fmaxf(md, sv);
      }
    }
    red[t] = md;                                // index = c*32 + r
  }
  __syncthreads();
  if (t < 32) {
    float md = red[t];
#pragma unroll
    for (int c = 1; c < 8; ++c) md = fmaxf(md, red[t + c * 32]);
    float ss = ssrc[row0 + t];
    ssl[t] = ss;
    float mm = 0.f;
    if (md > -1.0e37f) {
      float sc = ss + md;
      mm = sc > 0.f ? sc : NEG * sc;            // m = leakyrelu(max score)
    }
    mrow[t] = mm;
  }
  __syncthreads();

  // ---- phase 2: j-tiles of 256; bf16 P tile in LDS, WMMA against ht ----
  const int w  = t >> 5;                 // wave = N-tile (cols 16w..16w+15)
  const int L  = t & 31;
  const int r  = L & 15;
  const int lh = L >> 4;
  const float ss0 = ssl[r],      m0 = mrow[r];        // row r      (M-tile 0)
  const float ss1 = ssl[r + 16], m1 = mrow[r + 16];   // row r+16   (M-tile 1)
  const int coln = w * 16 + r;           // B col n = L&15 (== r)
  const __bf16* htc = ht + (size_t)coln * NN;
  const unsigned int* mw0 = mb + (size_t)(row0 + r) * (NN / 32);
  const unsigned int* mw1 = mb + (size_t)(row0 + r + 16) * (NN / 32);

  float zacc0 = 0.f, zacc1 = 0.f;
  v8f acc0 = {0.f,0.f,0.f,0.f,0.f,0.f,0.f,0.f};
  v8f acc1 = {0.f,0.f,0.f,0.f,0.f,0.f,0.f,0.f};

  for (int j0 = 0; j0 < NN; j0 += 256) {
    // prefetch next tile's B-fragment cachelines for this column
    if (j0 + 256 < NN) {
#pragma unroll
      for (int pf = 0; pf < 4; ++pf)
        __builtin_prefetch(htc + j0 + 256 + pf * 64, 0, 1);
    }
    // each lane: 16 probabilities for rows r and r+16, cols [jbase, jbase+16)
    int jbase = j0 + w * 32 + lh * 16;
    unsigned int bits0 = (mw0[jbase >> 5] >> (jbase & 16)) & 0xFFFFu;
    unsigned int bits1 = (mw1[jbase >> 5] >> (jbase & 16)) & 0xFFFFu;
    v16bf pv0, pv1;
#pragma unroll
    for (int b = 0; b < 16; ++b) {
      float sd = sdl[jbase + b];
      float sc0 = ss0 + sd, sc1 = ss1 + sd;
      float lr0 = sc0 > 0.f ? sc0 : NEG * sc0;
      float lr1 = sc1 > 0.f ? sc1 : NEG * sc1;
      float p0 = (bits0 & (1u << b)) ? __expf(lr0 - m0) : 0.f;
      float p1 = (bits1 & (1u << b)) ? __expf(lr1 - m1) : 0.f;
      zacc0 += p0; zacc1 += p1;
      pv0[b] = (__bf16)p0;
      pv1[b] = (__bf16)p1;
    }
    *(v16bf*)&ptile[r * 256 + w * 32 + lh * 16]        = pv0;   // 32B LDS store
    *(v16bf*)&ptile[(r + 16) * 256 + w * 32 + lh * 16] = pv1;
    __syncthreads();

#pragma unroll
    for (int sub = 0; sub < 8; ++sub) {
      // B 32x16 bf16 frag: lane n = L&15, K = lh*16 + e (contiguous in ht)
      v16bf bfr = *(const v16bf*)(htc + j0 + sub * 32 + lh * 16);
      // A 16x32 bf16 frags (per M-tile): lane half -> K {lh*16..lh*16+15}
      union { v16bf v; struct { v4u lo; v4u hi; } p; } af0, af1;
      int base0 = r * 256 + sub * 32 + lh * 16;
      af0.p.lo = *(const v4u*)&ptile[base0];
      af0.p.hi = *(const v4u*)&ptile[base0 + 8];
      int base1 = (r + 16) * 256 + sub * 32 + lh * 16;
      af1.p.lo = *(const v4u*)&ptile[base1];
      af1.p.hi = *(const v4u*)&ptile[base1 + 8];
      acc0 = __builtin_amdgcn_wmma_f32_16x16x32_bf16(false, af0.v, false, bfr,
                                                     (short)0, acc0, false, false);
      acc1 = __builtin_amdgcn_wmma_f32_16x16x32_bf16(false, af1.v, false, bfr,
                                                     (short)0, acc1, false, false);
    }
    __syncthreads();
  }

  // ---- Z reduction: 16 contributors per row (lanes with L&15==r, all waves)
  red[t]       = zacc0;
  red[256 + t] = zacc1;
  __syncthreads();
  if (t < 32) {
    int mt16 = t >> 4, rr = t & 15;
    float z = 0.f;
#pragma unroll
    for (int q = 0; q < 16; ++q) z += red[mt16 * 256 + rr + q * 16];
    Zl[t] = z;
  }
  __syncthreads();

  // ---- epilogue: out = O/Z + bias ; C/D layout lane n=L&15, VGPR v -> m=v+8*lh
  float bb = bias[coln];
#pragma unroll
  for (int v = 0; v < 8; ++v) {
    int m = v + lh * 8;
    out[(size_t)(row0 + m) * FOUT + coln]      = acc0[v] / Zl[m]      + bb;
    out[(size_t)(row0 + 16 + m) * FOUT + coln] = acc1[v] / Zl[m + 16] + bb;
  }
}

// ============================================================================
extern "C" void kernel_launch(void* const* d_in, const int* in_sizes, int n_in,
                              void* d_out, int out_size, void* d_ws, size_t ws_size,
                              hipStream_t stream) {
  const float* x    = (const float*)d_in[0];
  const int*   adj  = (const int*)  d_in[1];
  const float* W    = (const float*)d_in[2];
  const float* a1   = (const float*)d_in[3];
  const float* a2   = (const float*)d_in[4];
  const float* bias = (const float*)d_in[5];
  float* out = (float*)d_out;

  char* ws = (char*)d_ws;
  float*        h     = (float*)ws;                                  // 4 MB
  __bf16*       ht    = (__bf16*)(ws + ((size_t)4 << 20));           // 2 MB
  float*        ssrc  = (float*)(ws + ((size_t)6 << 20));            // 32 KB
  float*        sdst  = (float*)(ws + ((size_t)6 << 20) + (32 << 10));
  unsigned int* mbits = (unsigned int*)(ws + ((size_t)6 << 20) + (64 << 10)); // 8 MB

  gemm1_xw<<<NN / 16, 256, 0, stream>>>(x, W, h, ht);
  scores<<<NN / 8, 256, 0, stream>>>(h, a1, a2, ssrc, sdst);
  packmask<<<(size_t)NN * NN / 256, 256, 0, stream>>>(adj, mbits);
  gat_softmax_agg<<<NN / 32, 256, 0, stream>>>(ht, mbits, ssrc, sdst, bias, out);
}
